// RelativePositionMultiHeadAttention_79809082294654
// MI455X (gfx1250) — matrix operand order repaired and checked by reading, not compiled, not run on hardware
//
#include <hip/hip_runtime.h>

typedef __attribute__((ext_vector_type(16))) __bf16 v16bf;
typedef __attribute__((ext_vector_type(8)))  float  v8f;

#define B_   2
#define Hh   64
#define Ww   64
#define Cc   128
#define NH   4
#define KD   32
#define DV   32
#define Ll   4096      /* H*W */
#define BHL  32768     /* B*NH*L */
#define RL   127       /* 2*W-1 */

// ---------- helpers ----------
__device__ __forceinline__ unsigned short f2bf(float f) {
  unsigned u = __float_as_uint(f);
  unsigned r = u + 0x7FFFu + ((u >> 16) & 1u);   // round-to-nearest-even
  return (unsigned short)(r >> 16);
}
__device__ __forceinline__ float bf2f(unsigned short h) {
  return __uint_as_float(((unsigned)h) << 16);
}
// build a 16xbf16 WMMA fragment from two 16B chunks
__device__ __forceinline__ v16bf frag_ld(const unsigned short* p0, const unsigned short* p1) {
  union { uint4 u[2]; v16bf v; } r;
  r.u[0] = *(const uint4*)p0;
  r.u[1] = *(const uint4*)p1;
  return r.v;
}

// ---------- prologue: f32 -> bf16 packing ----------
__global__ void k_cvt_x(const float* __restrict__ x, unsigned short* __restrict__ xb, int n) {
  int i = blockIdx.x * 256 + threadIdx.x;
  if (i < n) xb[i] = f2bf(x[i]);
}
__global__ void k_cvt_w(const float* __restrict__ w, unsigned short* __restrict__ wT) {
  int i = blockIdx.x * 256 + threadIdx.x;
  if (i < Cc * 384) {
    int kk = i / 384, col = i % 384;
    wT[col * Cc + kk] = f2bf(w[i]);   // transpose so B-operand K-runs are contiguous
  }
}

// ---------- QKV projection: [8192,128]x[128,384], WMMA bf16 ----------
__global__ void __launch_bounds__(128) k_qkv(const unsigned short* __restrict__ xb,
                                             const unsigned short* __restrict__ wT,
                                             unsigned short* __restrict__ Qb,
                                             unsigned short* __restrict__ Kb,
                                             unsigned short* __restrict__ VTb) {
  int lane = threadIdx.x & 31, wv = threadIdx.x >> 5;
  int tile = blockIdx.x * 4 + wv;               // 512*24 = 12288 tiles
  int mt = tile / 24, nt = tile % 24;
  int hh = lane >> 4, l16 = lane & 15;

  const unsigned short* arow = xb + (size_t)(mt * 16 + l16) * Cc;
  const unsigned short* brow = wT + (size_t)(nt * 16 + l16) * Cc;
  v8f acc = {};
#pragma unroll
  for (int c = 0; c < 4; ++c) {
    v16bf a = frag_ld(arow + c * 32 + 8 * hh, arow + c * 32 + 16 + 8 * hh);
    v16bf b = frag_ld(brow + c * 32 + 8 * hh, brow + c * 32 + 16 + 8 * hh);
    acc = __builtin_amdgcn_wmma_f32_16x16x32_bf16(false, a, false, b, (short)0, acc, false, false);
  }
#pragma unroll
  for (int r = 0; r < 8; ++r) {
    int grow = mt * 16 + r + 8 * hh;            // flat b*L + l
    int col  = nt * 16 + l16;
    int b = grow >> 12, l = grow & 4095;
    unsigned short v = f2bf(acc[r]);
    if (col < 128) {                            // Q: [b,h,l,kd]
      int h = col >> 5, d = col & 31;
      Qb[((size_t)(b * NH + h) * Ll + l) * KD + d] = v;
    } else if (col < 256) {                     // K: [b,h,l,kd]
      int c2 = col - 128; int h = c2 >> 5, d = c2 & 31;
      Kb[((size_t)(b * NH + h) * Ll + l) * KD + d] = v;
    } else {                                    // V transposed: [b,h,dv,l]
      int c2 = col - 256; int h = c2 >> 5, d = c2 & 31;
      VTb[((size_t)(b * NH + h) * DV + d) * Ll + l] = v;
    }
  }
}

// ---------- factored relative bias: qw/qh[l][m] = q_l . pos_emb[m] ----------
__global__ void k_bias(const unsigned short* __restrict__ Qb,
                       const float* __restrict__ pw, const float* __restrict__ ph,
                       float* __restrict__ qw, float* __restrict__ qh) {
  long long i = (long long)blockIdx.x * 256 + threadIdx.x;
  const long long tot = (long long)BHL * RL;
  if (i >= 2 * tot) return;
  int which = (i >= tot);
  if (which) i -= tot;
  int bhl = (int)(i / RL), m = (int)(i % RL);
  const unsigned short* q = Qb + (size_t)bhl * KD;
  const float* e = (which ? ph : pw) + m * KD;
  float s = 0.f;
#pragma unroll
  for (int d = 0; d < KD; ++d) s += bf2f(q[d]) * e[d];
  (which ? qh : qw)[(size_t)bhl * 128 + m] = s;
}

// ---------- fused flash attention with decomposed relative bias ----------
__global__ void __launch_bounds__(128) k_flash(const unsigned short* __restrict__ Qb,
                                               const unsigned short* __restrict__ Kb,
                                               const unsigned short* __restrict__ VTb,
                                               const float* __restrict__ qw,
                                               const float* __restrict__ qh,
                                               float* __restrict__ out) {
  __shared__ float qh_s[64 * 128];              // 32 KB: qh rows for this query block
  __shared__ unsigned short pbuf[4][16 * 64];   // 8 KB: per-wave P staging (C->A layout)

  const int bid = blockIdx.x;
  const int i   = bid & 63;                     // query image row
  const int bh  = bid >> 6;                     // b*4+h
  const int h   = bh & 3;
  const int b   = bh >> 2;
  const int tid = threadIdx.x;
  const int lane = tid & 31, wv = tid >> 5;
  const int hh = lane >> 4, l16 = lane & 15;
  const float scale = 0.17677669529663687f;     // 1/sqrt(32), applied to q.k only

  const int qrow0 = bh * Ll + i * 64;           // global flat row of query 0 of block

  for (int idx = tid; idx < 64 * 128; idx += 128)
    qh_s[idx] = qh[(size_t)(qrow0 + (idx >> 7)) * 128 + (idx & 127)];
  __syncthreads();

  // Q A-operand fragment (loop invariant)
  const unsigned short* qrow = Qb + (size_t)(qrow0 + wv * 16 + l16) * KD;
  v16bf qa = frag_ld(qrow + 8 * hh, qrow + 16 + 8 * hh);

  // qw bias slice in registers, laid out to mirror the WMMA C layout:
  // element (t,r) of lane -> query m = wv*16+r+8*hh, key col j' = l16+16t,
  // bias index = (W-1) - j_m + j' = 63 - m + j'  (always in [0,126])
  float qwv[32];
#pragma unroll
  for (int t = 0; t < 4; ++t)
#pragma unroll
    for (int r = 0; r < 8; ++r) {
      int m = wv * 16 + r + 8 * hh;
      qwv[t * 8 + r] = qw[(size_t)(qrow0 + m) * 128 + (63 - m + l16 + 16 * t)];
    }

  float mrow[8], lrow[8];
#pragma unroll
  for (int r = 0; r < 8; ++r) { mrow[r] = -3.0e38f; lrow[r] = 0.f; }
  v8f oacc0 = {}, oacc1 = {};

  const unsigned short* kbase = Kb + (size_t)bh * Ll * KD;
  const unsigned short* vbase = VTb + (size_t)bh * DV * Ll;

  for (int n = 0; n < 64; ++n) {                // key tile = image row i' = n
    if (n + 1 < 64) {                           // global_prefetch_b8 next K tile
      __builtin_prefetch(kbase + (size_t)((n + 1) * 64 + lane) * KD, 0, 0);
      __builtin_prefetch(kbase + (size_t)((n + 1) * 64 + 32 + lane) * KD, 0, 0);
    }
    // qh term: index 63+n-i is uniform across the whole tile
    float qhv[8];
#pragma unroll
    for (int r = 0; r < 8; ++r)
      qhv[r] = qh_s[(wv * 16 + r + 8 * hh) * 128 + (63 + n - i)];

    // S = Q.K^T * scale + qw + qh   (4 N-tiles of 16 keys, one K=32 WMMA each)
    float sv[32];
#pragma unroll
    for (int t = 0; t < 4; ++t) {
      const unsigned short* krow = kbase + (size_t)(n * 64 + 16 * t + l16) * KD;
      v16bf kb = frag_ld(krow + 8 * hh, krow + 16 + 8 * hh);
      v8f z = {};
      v8f s = __builtin_amdgcn_wmma_f32_16x16x32_bf16(false, qa, false, kb, (short)0, z, false, false);
#pragma unroll
      for (int r = 0; r < 8; ++r)
        sv[t * 8 + r] = s[r] * scale + qwv[t * 8 + r] + qhv[r];
    }

    // online softmax: rows live in 16-lane halves -> xor-butterfly over 1,2,4,8
    float alpha[8];
#pragma unroll
    for (int r = 0; r < 8; ++r) {
      float rm = sv[r];
#pragma unroll
      for (int t = 1; t < 4; ++t) rm = fmaxf(rm, sv[t * 8 + r]);
      rm = fmaxf(rm, __shfl_xor(rm, 1, 32));
      rm = fmaxf(rm, __shfl_xor(rm, 2, 32));
      rm = fmaxf(rm, __shfl_xor(rm, 4, 32));
      rm = fmaxf(rm, __shfl_xor(rm, 8, 32));
      float mn = fmaxf(mrow[r], rm);
      alpha[r] = __expf(mrow[r] - mn);
      mrow[r] = mn;
      float rs = 0.f;
#pragma unroll
      for (int t = 0; t < 4; ++t) {
        float p = __expf(sv[t * 8 + r] - mn);
        sv[t * 8 + r] = p;
        rs += p;
      }
      rs += __shfl_xor(rs, 1, 32);
      rs += __shfl_xor(rs, 2, 32);
      rs += __shfl_xor(rs, 4, 32);
      rs += __shfl_xor(rs, 8, 32);
      lrow[r] = lrow[r] * alpha[r] + rs;
      oacc0[r] *= alpha[r];
      oacc1[r] *= alpha[r];
    }

    // P (C layout) -> LDS row-major 16x64 bf16, then re-read as A-operand.
    // Per-wave private buffer; DS ops are in-order within a wave.
#pragma unroll
    for (int t = 0; t < 4; ++t)
#pragma unroll
      for (int r = 0; r < 8; ++r)
        pbuf[wv][(r + 8 * hh) * 64 + l16 + 16 * t] = f2bf(sv[t * 8 + r]);
    __builtin_amdgcn_wave_barrier();

    // O += P @ V  (K=64 split into two K=32 WMMAs; V^T gives contiguous B frags)
#pragma unroll
    for (int kh = 0; kh < 2; ++kh) {
      const unsigned short* prow = &pbuf[wv][l16 * 64 + 32 * kh];
      v16bf pa = frag_ld(prow + 8 * hh, prow + 16 + 8 * hh);
      const unsigned short* vr0 = vbase + (size_t)(l16)      * Ll + n * 64 + kh * 32;
      const unsigned short* vr1 = vbase + (size_t)(16 + l16) * Ll + n * 64 + kh * 32;
      v16bf vb0 = frag_ld(vr0 + 8 * hh, vr0 + 16 + 8 * hh);
      v16bf vb1 = frag_ld(vr1 + 8 * hh, vr1 + 16 + 8 * hh);
      oacc0 = __builtin_amdgcn_wmma_f32_16x16x32_bf16(false, pa, false, vb0, (short)0, oacc0, false, false);
      oacc1 = __builtin_amdgcn_wmma_f32_16x16x32_bf16(false, pa, false, vb1, (short)0, oacc1, false, false);
    }
  }

  // epilogue: out[b, i, j=m, c = h*32+dv] = O / l
#pragma unroll
  for (int r = 0; r < 8; ++r) {
    int m = wv * 16 + r + 8 * hh;
    float inv = 1.f / lrow[r];
    size_t base = (size_t)(b * Ll + i * 64 + m) * Cc + h * DV;
    out[base + l16]      = oacc0[r] * inv;
    out[base + 16 + l16] = oacc1[r] * inv;
  }
}

// ---------- launch ----------
extern "C" void kernel_launch(void* const* d_in, const int* in_sizes, int n_in,
                              void* d_out, int out_size, void* d_ws, size_t ws_size,
                              hipStream_t stream) {
  (void)in_sizes; (void)n_in; (void)out_size; (void)ws_size;
  const float* x  = (const float*)d_in[0];   // [2,64,64,128]
  const float* w  = (const float*)d_in[1];   // [128,384]
  const float* ph = (const float*)d_in[2];   // [127,32]
  const float* pw = (const float*)d_in[3];   // [127,32]
  float* out = (float*)d_out;

  char* ws = (char*)d_ws;
  size_t off = 0;
  auto take = [&](size_t bytes) {
    void* p = ws + off;
    off += (bytes + 255) & ~(size_t)255;
    return p;
  };
  unsigned short* xb  = (unsigned short*)take((size_t)8192 * 128 * 2);   // 2 MB
  unsigned short* wT  = (unsigned short*)take((size_t)384 * 128 * 2);    // 96 KB
  unsigned short* Qb  = (unsigned short*)take((size_t)BHL * KD * 2);     // 2 MB
  unsigned short* Kb  = (unsigned short*)take((size_t)BHL * KD * 2);     // 2 MB
  unsigned short* VTb = (unsigned short*)take((size_t)BHL * DV * 2);     // 2 MB
  float* qwb = (float*)take((size_t)BHL * 128 * 4);                      // 16 MB
  float* qhb = (float*)take((size_t)BHL * 128 * 4);                      // 16 MB

  k_cvt_x<<<4096, 256, 0, stream>>>(x, xb, B_ * Ll * Cc);
  k_cvt_w<<<192, 256, 0, stream>>>(w, wT);
  k_qkv<<<3072, 128, 0, stream>>>(xb, wT, Qb, Kb, VTb);
  const long long nbias = 2LL * BHL * RL;               // 8,323,072
  k_bias<<<(int)((nbias + 255) / 256), 256, 0, stream>>>(Qb, pw, ph, qwb, qhb);
  k_flash<<<B_ * NH * (Ll / 64), 128, 0, stream>>>(Qb, Kb, VTb, qwb, qhb, out);
}
